// RNN_88811333747116
// MI455X (gfx1250) — matrix operand order repaired and checked by reading, not compiled, not run on hardware
//
#include <hip/hip_runtime.h>
#include <hip/hip_bf16.h>
#include <math.h>

#define B_SZ   32
#define T_SZ   1024
#define D_SZ   512
#define H_SZ   512
#define G_SZ   2048      // 4*H
#define NWG    16
#define CPW    32        // h-columns owned per workgroup (H_SZ / NWG)
#define USTR   520       // padded LDS row stride (elements) -> conflict-free ds_load_b128

typedef __attribute__((ext_vector_type(16))) __bf16 bf16x16;
typedef __attribute__((ext_vector_type(8)))  __bf16 bf16x8;
typedef __attribute__((ext_vector_type(8)))  float  f32x8;

// ---- A fragment loader: 16x32 bf16 A-matrix, row-major source ----
// contiguous K order (matches B): lane half 0 -> K {0..15}, half 1 -> K {16..31}
__device__ __forceinline__ bf16x16 load_a_frag(const __bf16* __restrict__ p, int half) {
  bf16x8 c0 = *(const bf16x8*)(p + (half << 4));
  bf16x8 c1 = *(const bf16x8*)(p + 8 + (half << 4));
  bf16x16 a;
#pragma unroll
  for (int i = 0; i < 8; ++i) { a[i] = c0[i]; a[i + 8] = c1[i]; }
  return a;
}

__device__ __forceinline__ f32x8 wmma_bf16(bf16x16 a, bf16x16 b, f32x8 c) {
  // (neg_a, A, neg_b, B, c_mod, C, reuse_a, reuse_b)
  return __builtin_amdgcn_wmma_f32_16x16x32_bf16(false, a, false, b, (short)0, c, false, false);
}

__device__ __forceinline__ float sigmoidf_(float x) {
  return 1.0f / (1.0f + __expf(-x));
}

// ---------------- pre-pass kernels ----------------
__global__ void cvt_bf16(const float* __restrict__ src, __bf16* __restrict__ dst, int n) {
  int i = blockIdx.x * blockDim.x + threadIdx.x;
  if (i < n) dst[i] = (__bf16)src[i];
}

// src [rows][cols] fp32 row-major -> dst [cols][rows] bf16 (transposed)
__global__ void transpose_bf16(const float* __restrict__ src, __bf16* __restrict__ dst,
                               int rows, int cols) {
  int i = blockIdx.x * blockDim.x + threadIdx.x;
  if (i < rows * cols) {
    int r = i / cols, c = i % cols;
    dst[(size_t)c * rows + r] = (__bf16)src[i];
  }
}

__global__ void init_state(__bf16* __restrict__ hb, unsigned* __restrict__ bar) {
  int i = blockIdx.x * blockDim.x + threadIdx.x;
  if (i < B_SZ * H_SZ) hb[i] = (__bf16)0.0f;
  if (i == 0) { bar[0] = 0u; bar[1] = 0u; }
}

// ---------------- persistent LSTM kernel ----------------
__device__ __forceinline__ void grid_sync(unsigned* __restrict__ cnt,
                                          unsigned* __restrict__ gen) {
  __threadfence();           // release this thread's h stores to device scope
  __syncthreads();
  if (threadIdx.x == 0) {
    unsigned g = __hip_atomic_load(gen, __ATOMIC_RELAXED, __HIP_MEMORY_SCOPE_AGENT);
    unsigned a = __hip_atomic_fetch_add(cnt, 1u, __ATOMIC_ACQ_REL, __HIP_MEMORY_SCOPE_AGENT);
    if (a == NWG - 1) {
      __hip_atomic_store(cnt, 0u, __ATOMIC_RELAXED, __HIP_MEMORY_SCOPE_AGENT);
      __hip_atomic_fetch_add(gen, 1u, __ATOMIC_RELEASE, __HIP_MEMORY_SCOPE_AGENT);
    } else {
      while (__hip_atomic_load(gen, __ATOMIC_ACQUIRE, __HIP_MEMORY_SCOPE_AGENT) == g)
        __builtin_amdgcn_s_sleep(2);
    }
  }
  __syncthreads();
  __threadfence();           // acquire other blocks' h stores
}

__global__ __launch_bounds__(256, 1) void lstm_persist(
    const __bf16* __restrict__ xb,   // [B][T][D] bf16
    const __bf16* __restrict__ Wt,   // [4H][D]  bf16 (transposed)
    const __bf16* __restrict__ Ut,   // [4H][H]  bf16 (transposed)
    const float*  __restrict__ bias, // [4H] fp32
    __bf16*       __restrict__ hb,   // [B][H] bf16 (recurrent state)
    float*        __restrict__ out,  // [B][T][H] fp32
    unsigned*     __restrict__ bar)  // {count, generation}
{
  // U-slice for this WG, LDS-resident for all 1024 steps (recurrent critical path).
  __shared__ __bf16 ulds[4 * CPW * USTR];      // 128 x 520 bf16 = 130 KB
  __shared__ float  zbuf[B_SZ * 4 * CPW];      // 32 x 128 f32   =  16 KB

  const int tid  = threadIdx.x;
  const int wg   = blockIdx.x;       // 0..15: owns h-cols [wg*32, wg*32+32)
  const int wv   = tid >> 5;         // wave 0..7
  const int lane = tid & 31;
  const int gate = wv >> 1;          // 0..3 (i,f,g,o)
  const int mh   = wv & 1;           // M-half of the 32-row batch
  const int nl   = lane & 15;
  const int half = lane >> 4;

  // ---- fill LDS with this WG's 128 U-columns (16B chunks, coalesced) ----
  for (int idx = tid; idx < (4 * CPW) * (D_SZ / 8); idx += 256) {
    const int lc  = idx >> 6;            // local col 0..127
    const int c16 = idx & 63;            // 16B chunk within the 1KB row
    const int g   = lc >> 5, j = lc & 31;
    const size_t grow = (size_t)(g * H_SZ + wg * CPW + j) * H_SZ;
    *(uint4*)(ulds + lc * USTR + c16 * 8) = *(const uint4*)(Ut + grow + c16 * 8);
  }

  const int    batch = mh * 16 + nl;                   // A row for this lane
  const __bf16* hRow = hb + batch * H_SZ;
  const size_t xRow  = (size_t)batch * T_SZ * D_SZ;

  const int colBase = gate * H_SZ + wg * CPW;          // global gate-column base
  const __bf16* w0 = Wt + (size_t)(colBase +      nl) * D_SZ + (half << 4);
  const __bf16* w1 = Wt + (size_t)(colBase + 16 + nl) * D_SZ + (half << 4);
  const __bf16* u0 = ulds + (gate * CPW +      nl) * USTR + (half << 4);
  const __bf16* u1 = ulds + (gate * CPW + 16 + nl) * USTR + (half << 4);

  // elementwise mapping: element e = q*256 + tid -> row = q*8 + tid>>5, jl = tid&31
  const int jl   = tid & 31;
  const int gcol = wg * CPW + jl;
  const float b_i = bias[0 * H_SZ + gcol];
  const float b_f = bias[1 * H_SZ + gcol];
  const float b_g = bias[2 * H_SZ + gcol];
  const float b_o = bias[3 * H_SZ + gcol];

  float cst[4];
#pragma unroll
  for (int q = 0; q < 4; ++q) cst[q] = 0.0f;

  // x_t @ W stage (independent of h -> software-pipelined across the grid barrier)
  auto do_xw = [&](int t, f32x8& p0, f32x8& p1) {
    const __bf16* aX = xb + xRow + (size_t)t * D_SZ;
    __builtin_prefetch(aX + 2 * D_SZ, 0, 0);   // global_prefetch for t+2's row
#pragma unroll 4
    for (int k = 0; k < D_SZ; k += 32) {
      bf16x16 a = load_a_frag(aX + k, half);
      p0 = wmma_bf16(a, *(const bf16x16*)(w0 + k), p0);
      p1 = wmma_bf16(a, *(const bf16x16*)(w1 + k), p1);
    }
  };

  __syncthreads();   // ulds ready

  f32x8 pw0 = {0.f, 0.f, 0.f, 0.f, 0.f, 0.f, 0.f, 0.f};
  f32x8 pw1 = {0.f, 0.f, 0.f, 0.f, 0.f, 0.f, 0.f, 0.f};
  do_xw(0, pw0, pw1);

  for (int t = 0; t < T_SZ; ++t) {
    // ---- z = (x_t @ W) + h_{t-1} @ U ;  U from LDS (critical path) ----
    f32x8 acc0 = pw0, acc1 = pw1;
#pragma unroll 4
    for (int k = 0; k < H_SZ; k += 32) {
      bf16x16 a = load_a_frag(hRow + k, half);
      acc0 = wmma_bf16(a, *(const bf16x16*)(u0 + k), acc0);
      acc1 = wmma_bf16(a, *(const bf16x16*)(u1 + k), acc1);
    }

    // ---- scatter accumulators to LDS z-buffer (C layout per ISA 7.12.2) ----
    {
      const int zr = mh * 16 + half * 8;
      const int zc = gate * CPW + nl;
#pragma unroll
      for (int r = 0; r < 8; ++r) {
        zbuf[(zr + r) * (4 * CPW) + zc]      = acc0[r];
        zbuf[(zr + r) * (4 * CPW) + zc + 16] = acc1[r];
      }
    }
    __syncthreads();

    // ---- gates + state update; c stays in registers ----
#pragma unroll
    for (int q = 0; q < 4; ++q) {
      const int row = q * 8 + (tid >> 5);
      const float* zrow = zbuf + row * (4 * CPW);
      float zi = zrow[0 * CPW + jl] + b_i;
      float zf = zrow[1 * CPW + jl] + b_f;
      float zg = zrow[2 * CPW + jl] + b_g;
      float zo = zrow[3 * CPW + jl] + b_o;
      float ig = sigmoidf_(zi);
      float fg = sigmoidf_(zf);
      float gg = tanhf(zg);
      float og = sigmoidf_(zo);
      float cn = fg * cst[q] + ig * gg;
      cst[q]   = cn;
      float hn = og * tanhf(cn);
      const int col = wg * CPW + jl;
      hb[row * H_SZ + col] = (__bf16)hn;                        // next step's A operand
      out[((size_t)row * T_SZ + t) * H_SZ + col] = hn;          // [B][T][H] fp32
    }

    // ---- overlap x_{t+1}@W with the barrier wait (off the critical path) ----
    pw0 = (f32x8){0.f, 0.f, 0.f, 0.f, 0.f, 0.f, 0.f, 0.f};
    pw1 = pw0;
    if (t + 1 < T_SZ) do_xw(t + 1, pw0, pw1);

    grid_sync(bar, bar + 1);
  }
}

// ---------------- host launcher ----------------
extern "C" void kernel_launch(void* const* d_in, const int* in_sizes, int n_in,
                              void* d_out, int out_size, void* d_ws, size_t ws_size,
                              hipStream_t stream) {
  (void)in_sizes; (void)n_in; (void)out_size; (void)ws_size;
  const float* x = (const float*)d_in[0];   // [32][1024][512]
  const float* W = (const float*)d_in[1];   // [512][2048]
  const float* U = (const float*)d_in[2];   // [512][2048]
  const float* b = (const float*)d_in[3];   // [2048]
  float* out = (float*)d_out;

  char* ws = (char*)d_ws;
  __bf16*   xb  = (__bf16*)(ws);                        // 33,554,432 B
  __bf16*   Wt  = (__bf16*)(ws + (size_t)33554432);     //  2,097,152 B
  __bf16*   Ut  = (__bf16*)(ws + (size_t)35651584);     //  2,097,152 B
  __bf16*   hb  = (__bf16*)(ws + (size_t)37748736);     //     32,768 B
  unsigned* bar = (unsigned*)(ws + (size_t)37781504);   // {count, gen}

  const int nx = B_SZ * T_SZ * D_SZ;
  hipLaunchKernelGGL(cvt_bf16, dim3((nx + 255) / 256), dim3(256), 0, stream, x, xb, nx);
  const int nw = D_SZ * G_SZ;
  hipLaunchKernelGGL(transpose_bf16, dim3((nw + 255) / 256), dim3(256), 0, stream, W, Wt, D_SZ, G_SZ);
  hipLaunchKernelGGL(transpose_bf16, dim3((nw + 255) / 256), dim3(256), 0, stream, U, Ut, D_SZ, G_SZ);
  hipLaunchKernelGGL(init_state, dim3(64), dim3(256), 0, stream, hb, bar);
  hipLaunchKernelGGL(lstm_persist, dim3(NWG), dim3(256), 0, stream,
                     xb, Wt, Ut, b, hb, out, bar);
}
